// FFTConv1d_3143916060960
// MI455X (gfx1250) — matrix operand order repaired and checked by reading, not compiled, not run on hardware
//
#include <hip/hip_runtime.h>

// ---------------- problem constants ----------------
#define BATCH   8
#define CI      256
#define CO      256
#define LEN     8192
#define KW      4096
#define PADL    2047
#define XSTRIDE 12288                        // LEN + KW - 1 = 12287, rounded up
#define XTOTAL  (BATCH * CI * XSTRIDE)       // 25,165,824 halfs (~48 MB)
#define XSLACK  512                          // zero tail so strip loads never stray

// ---------------- GEMM tiling ----------------
// M-side = l (Toeplitz strip from Xpad), N-side = o (weight)
#define LT   256   // l per workgroup
#define OT   128   // o per workgroup
#define KC   32    // reduction chunk (one WMMA k-step)
#define NITER (CI * (KW / KC))     // 32768
#define NPAIR 144                  // strip pair-array entries (covers t<=287)

typedef __attribute__((ext_vector_type(16))) _Float16 v16h;
typedef __attribute__((ext_vector_type(8)))  float    v8f;

// -------- kernel 1: bake 'same' padding, fp32 -> f16 --------
__global__ __launch_bounds__(256) void pad_x_kernel(const float* __restrict__ x,
                                                    _Float16* __restrict__ xpad) {
    unsigned int idx = blockIdx.x * 256u + threadIdx.x;
    if (idx >= (unsigned int)(XTOTAL + XSLACK)) return;
    if (idx >= (unsigned int)XTOTAL) { xpad[idx] = (_Float16)0.0f; return; }
    unsigned int row = idx / XSTRIDE;        // b*CI + i
    int t  = (int)(idx - row * XSTRIDE);
    int ls = t - PADL;
    float v = (ls >= 0 && ls < LEN) ? x[(size_t)row * LEN + ls] : 0.0f;
    xpad[idx] = (_Float16)v;
}

// -------- kernel 2: implicit-GEMM conv via V_WMMA_F32_16X16X32_F16 --------
// Wave tile: 64 l x 64 o = 4x4 subtiles = 16 WMMA per k-step.
// Double-buffered LDS: one barrier per k-step, global prefetch 2 steps ahead.
__global__ __launch_bounds__(256) void conv_wmma_kernel(const float* __restrict__ w,
                                                        const float* __restrict__ bias,
                                                        const _Float16* __restrict__ xpad,
                                                        float* __restrict__ out) {
    __shared__ __align__(16) _Float16 Ws[2][OT * KC];   // weight tiles f16 (2 x 8 KB)
    __shared__ unsigned int pe[2][NPAIR + 8];           // strip even pairs
    __shared__ unsigned int po[2][NPAIR + 8];           // strip odd  pairs

    const int tid  = threadIdx.x;
    const int lane = tid & 31;
    const int wave = tid >> 5;            // 0..7
    const int wl   = (wave & 3) * 64;     // wave l-offset (4 slices)
    const int wo   = (wave >> 2) * 64;    // wave o-offset (2 slices)
    const int n    = lane & 15;
    const int h    = lane >> 4;
    const int par  = n & 1;

    const int o0 = blockIdx.x * OT;               // 2 o-tiles
    const int tn = blockIdx.y;                    // 256 l-tiles
    const int b  = tn >> 5;                       // 32 l-tiles per batch
    const int l0 = (tn & 31) * LT;

    // staging roles: weight 128 rows x 32 k, 16 floats/thread
    const int ow = tid >> 1;                      // 0..127
    const int kh = (tid & 1) * 16;                // 0 or 16
    const bool do_strip = (tid < NPAIR);

    const size_t wbase = ((size_t)(o0 + ow) * CI) * KW + kh;
    const size_t xbase = ((size_t)b * CI) * XSTRIDE + (size_t)l0;

    v8f acc[16];
    #pragma unroll
    for (int s = 0; s < 16; ++s)
        #pragma unroll
        for (int v = 0; v < 8; ++v) acc[s][v] = 0.0f;

    float4 fr[4];
    unsigned int sa = 0, sb = 0;

    auto load_glb = [&](int it) {
        const int i  = it >> 7;                   // KW/KC = 128 chunks per i
        const int k0 = (it & 127) << 5;
        const float* src = w + wbase + (size_t)i * KW + k0;
        fr[0] = ((const float4*)src)[0];
        fr[1] = ((const float4*)src)[1];
        fr[2] = ((const float4*)src)[2];
        fr[3] = ((const float4*)src)[3];
        if (do_strip) {
            const unsigned int* xs32 =
                (const unsigned int*)(xpad + xbase + (size_t)i * XSTRIDE + k0);
            sa = xs32[tid];
            sb = xs32[tid + 1];
        }
    };

    auto commit = [&](int buf) {
        union { _Float16 hh[16]; uint4 q[2]; } pk;
        #pragma unroll
        for (int t = 0; t < 4; ++t) {
            pk.hh[4*t+0] = (_Float16)fr[t].x; pk.hh[4*t+1] = (_Float16)fr[t].y;
            pk.hh[4*t+2] = (_Float16)fr[t].z; pk.hh[4*t+3] = (_Float16)fr[t].w;
        }
        uint4* Wq = (uint4*)Ws[buf];              // 4 uint4 per 32-half row
        Wq[ow * 4 + (tid & 1) * 2 + 0] = pk.q[0];
        Wq[ow * 4 + (tid & 1) * 2 + 1] = pk.q[1];
        if (do_strip) {
            pe[buf][tid] = sa;
            po[buf][tid] = (sa >> 16) | (sb << 16);
        }
    };

    // ---- pipeline prologue ----
    load_glb(0);
    commit(0);
    __syncthreads();
    load_glb(1);

    for (int it = 0; it < NITER; ++it) {
        const int cur = it & 1;

        // ---- build fragments from current buffers ----
        union AF { v16h v; unsigned int u[8]; } af[4];   // Toeplitz x (A side)
        union BF { v16h v; uint4 q[2]; } bf[4];          // weight (B side)
        const unsigned int* pairs = par ? po[cur] : pe[cur];
        #pragma unroll
        for (int sm = 0; sm < 4; ++sm) {
            const int pb = (wl + sm * 16 + n - par) >> 1;
            #pragma unroll
            for (int g = 0; g < 2; ++g)
                #pragma unroll
                for (int j = 0; j < 4; ++j)
                    af[sm].u[g * 4 + j] = pairs[pb + g * 8 + h * 4 + j];
        }
        const uint4* Wq = (const uint4*)Ws[cur];
        #pragma unroll
        for (int sn = 0; sn < 4; ++sn) {
            const int oc = wo + sn * 16 + n;             // o column within WG tile
            bf[sn].q[0] = Wq[oc * 4 + h];                // K = h*8 + 0..7
            bf[sn].q[1] = Wq[oc * 4 + 2 + h];            // K = 16 + h*8 + 0..7
        }

        // ---- 16 WMMAs ----
        #pragma unroll
        for (int sm = 0; sm < 4; ++sm)
            #pragma unroll
            for (int sn = 0; sn < 4; ++sn)
                acc[sm * 4 + sn] = __builtin_amdgcn_wmma_f32_16x16x32_f16(
                    false, af[sm].v, false, bf[sn].v, (short)0,
                    acc[sm * 4 + sn], false, false);

        // ---- commit next chunk to other buffer, single barrier, refill regs ----
        if (it + 1 < NITER) {
            commit(cur ^ 1);
            __syncthreads();
            if (it + 2 < NITER) load_glb(it + 2);
            if (it + 4 < NITER) {
                const int i4 = (it + 4) >> 7;
                const int k4 = ((it + 4) & 127) << 5;
                __builtin_prefetch(w + wbase + (size_t)i4 * KW + k4, 0, 0);
            }
        }
    }

    // ---- epilogue: rows (h*8+v) are contiguous in l -> vector stores ----
    #pragma unroll
    for (int sn = 0; sn < 4; ++sn) {
        const int o  = o0 + wo + sn * 16 + n;
        const float bv = bias[o];
        #pragma unroll
        for (int sm = 0; sm < 4; ++sm) {
            const v8f a = acc[sm * 4 + sn];
            const size_t idx = ((size_t)b * CO + o) * LEN + l0 + wl + sm * 16 + h * 8;
            float4 r0 = { a[0] + bv, a[1] + bv, a[2] + bv, a[3] + bv };
            float4 r1 = { a[4] + bv, a[5] + bv, a[6] + bv, a[7] + bv };
            *(float4*)&out[idx]     = r0;
            *(float4*)&out[idx + 4] = r1;
        }
    }
}

extern "C" void kernel_launch(void* const* d_in, const int* in_sizes, int n_in,
                              void* d_out, int out_size, void* d_ws, size_t ws_size,
                              hipStream_t stream) {
    const float* x    = (const float*)d_in[0];
    const float* w    = (const float*)d_in[1];
    const float* bias = (const float*)d_in[2];
    float* out        = (float*)d_out;
    _Float16* xpad    = (_Float16*)d_ws;       // needs ~48 MB + slack

    const unsigned int totalPad = XTOTAL + XSLACK;
    pad_x_kernel<<<(totalPad + 255) / 256, 256, 0, stream>>>(x, xpad);
    conv_wmma_kernel<<<dim3(CO / OT, (BATCH * LEN) / LT), 256, 0, stream>>>(w, bias, xpad, out);
}
